// CoraGCN_88424786690103
// MI455X (gfx1250) — matrix operand order repaired and checked by reading, not compiled, not run on hardware
//
#include <hip/hip_runtime.h>

// ---------------------------------------------------------------------------
// 2-layer GCN for MI455X (gfx1250, wave32, WMMA).
// GEMMs use v_wmma_f32_16x16x32_bf16 with fragment-packed B (one b128-pair
// load per fragment); aggregation uses edge-parallel float atomics
// (bandwidth bound against 23.3 TB/s HBM).
// ---------------------------------------------------------------------------

typedef __attribute__((ext_vector_type(16))) __bf16       v16bf;
typedef __attribute__((ext_vector_type(8)))  float        v8f;
typedef __attribute__((ext_vector_type(8)))  unsigned int v8u;

#define GCN_IN  512
#define GCN_HID 128
#define GCN_CLS 32

// ---------------- fp32 -> bf16 conversion (grid-stride) ----------------
__global__ void k_f32_to_bf16(const float* __restrict__ in, __bf16* __restrict__ out,
                              size_t n) {
  size_t i = (size_t)blockIdx.x * blockDim.x + threadIdx.x;
  size_t stride = (size_t)gridDim.x * blockDim.x;
  for (; i < n; i += stride) out[i] = (__bf16)in[i];
}

// ---------------- pack fp32 weight [K, NCOLS] into WMMA B-fragment order ----
// Packed layout: dword index = ((kt*NT + nt)*32 + lane)*8 + v
//   where fragment (kt, nt), lane holds rows K = kt*32 + 2*(v&3)+16*(v>>2)+8*(lane>>4) (lo)
//   and +1 (hi), column N = nt*16 + (lane&15).
// Each lane then loads its whole fragment as 32 contiguous bytes.
__global__ void k_pack_w(const float* __restrict__ W, unsigned int* __restrict__ P,
                         int K, int NCOLS) {
  const int NT    = NCOLS / 16;
  const int total = (K / 32) * NT * 256;  // 32 lanes * 8 dwords per fragment
  int idx = blockIdx.x * blockDim.x + threadIdx.x;
  if (idx >= total) return;
  const int v    = idx & 7;
  const int lane = (idx >> 3) & 31;
  const int tile = idx >> 8;
  const int kt   = tile / NT;
  const int nt   = tile - kt * NT;
  const int hi   = lane >> 4;
  const int l    = lane & 15;
  const int krow = kt * 32 + 2 * (v & 3) + 16 * (v >> 2) + 8 * hi;
  const int col  = nt * 16 + l;
  const __bf16 lo = (__bf16)W[(size_t)krow * NCOLS + col];
  const __bf16 hb = (__bf16)W[(size_t)(krow + 1) * NCOLS + col];
  const unsigned int ulo = (unsigned int)__builtin_bit_cast(unsigned short, lo);
  const unsigned int uhb = (unsigned int)__builtin_bit_cast(unsigned short, hb);
  P[idx] = ulo | (uhb << 16);
}

// ---------------- fill float ----------------
__global__ void k_fill_f32(float* __restrict__ p, float v, size_t n) {
  size_t i = (size_t)blockIdx.x * blockDim.x + threadIdx.x;
  if (i < n) p[i] = v;
}

// ---------------- degree accumulation: deg[dst] += 1 ----------------
__global__ void k_degree(const int* __restrict__ dst, float* __restrict__ deg, int E) {
  int e = blockIdx.x * blockDim.x + threadIdx.x;
  if (e < E) atomicAdd(&deg[dst[e]], 1.0f);
}

// ---------------- dinv = deg>0 ? rsqrt(deg) : 0 ----------------
__global__ void k_dinv(const float* __restrict__ deg, float* __restrict__ dinv, int n) {
  int i = blockIdx.x * blockDim.x + threadIdx.x;
  if (i < n) {
    float d = deg[i];
    dinv[i] = (d > 0.0f) ? rsqrtf(d) : 0.0f;
  }
}

// ---------------- WMMA GEMM: C[M,NCOLS] = A[M,K] * Bpacked ----------------
// Row-major bf16 A, fragment-packed bf16 B (see k_pack_w), f32 C.
// One wave owns a 16-row x NCOLS strip. Per k-step: 2 b128 A loads +
// NT b128-pair B loads (one clause), then NT back-to-back WMMAs.
template <int NCOLS, int K>
__global__ void k_gemm_bf16(const __bf16* __restrict__ A, const v8u* __restrict__ Bp,
                            float* __restrict__ C, int M) {
  constexpr int NT = NCOLS / 16;  // N tiles
  constexpr int KT = K / 32;      // K steps

  const int wave = threadIdx.x >> 5;
  const int lane = threadIdx.x & 31;
  const int l    = lane & 15;  // column / row-within-half
  const int hi   = lane >> 4;  // half-wave select

  const int mtile = blockIdx.x * (blockDim.x >> 5) + wave;
  const int row0  = mtile * 16;
  if (row0 >= M) return;  // wave-uniform: EXEC stays all-ones for WMMA

  const int row = row0 + l;  // A row for this lane (M = 100000 is 16-aligned)

  v8f acc[NT];
  const v8f vzero = {0.f, 0.f, 0.f, 0.f, 0.f, 0.f, 0.f, 0.f};
#pragma unroll
  for (int i = 0; i < NT; ++i) acc[i] = vzero;

  // A row as 16-byte chunks (8 bf16 each); base is 16B aligned (K mult of 8).
  const uint4* ArowQ = (const uint4*)(A + (size_t)row * K);

#pragma unroll 1
  for (int kt = 0; kt < KT; ++kt) {
    const int k0 = kt * 32;

    // ---- B fragments: one v8u (32 B) per lane per fragment ----
    v8u bu[NT];
#pragma unroll
    for (int nt = 0; nt < NT; ++nt)
      bu[nt] = Bp[(size_t)(kt * NT + nt) * 32 + lane];

    // ---- A fragment: two runs of 8 consecutive bf16 ----
    // lanes 0-15: K k0+0..7 (VGPR0-3) and k0+16..23 (VGPR4-7); lanes 16-31: +8.
    const uint4 a0 = ArowQ[(k0 + 8 * hi) >> 3];
    const uint4 a1 = ArowQ[(k0 + 16 + 8 * hi) >> 3];
    const v8u au = {a0.x, a0.y, a0.z, a0.w, a1.x, a1.y, a1.z, a1.w};
    const v16bf afrag = __builtin_bit_cast(v16bf, au);

    // ---- WMMA burst ----
#pragma unroll
    for (int nt = 0; nt < NT; ++nt) {
      const v16bf bfrag = __builtin_bit_cast(v16bf, bu[nt]);
      acc[nt] = __builtin_amdgcn_wmma_f32_16x16x32_bf16(
          /*neg_a=*/false, afrag, /*neg_b=*/false, bfrag,
          /*c_mod=*/(short)0, acc[nt], /*reuse_a=*/false, /*reuse_b=*/false);
    }
  }

  // ---- store C: VGPR r holds (M = row0 + r + 8*hi, N = n0 + l) ----
#pragma unroll
  for (int nt = 0; nt < NT; ++nt) {
    const int n0 = nt * 16;
#pragma unroll
    for (int r = 0; r < 8; ++r) {
      const int m = row0 + r + 8 * hi;
      C[(size_t)m * NCOLS + n0 + l] = acc[nt][r];
    }
  }
}

// ---------------- edge scatter, 128 features: 32 lanes/edge x float4 ----------------
__global__ void k_scatter128(const int* __restrict__ src, const int* __restrict__ dst,
                             const float* __restrict__ dinv, const float* __restrict__ h,
                             float* __restrict__ out, int E) {
  const long long tid = (long long)blockIdx.x * blockDim.x + threadIdx.x;
  const long long e   = tid >> 5;
  const int lane      = (int)(tid & 31);
  if (e >= E) return;
  const int s = src[e], d = dst[e];
  const float nrm = dinv[s] * dinv[d];
  const float4 v = *(const float4*)(h + (size_t)s * GCN_HID + lane * 4);
  float* o = out + (size_t)d * GCN_HID + lane * 4;
  atomicAdd(o + 0, v.x * nrm);
  atomicAdd(o + 1, v.y * nrm);
  atomicAdd(o + 2, v.z * nrm);
  atomicAdd(o + 3, v.w * nrm);
}

// ---------------- edge scatter, 32 features: 8 lanes/edge x float4 ----------------
__global__ void k_scatter32(const int* __restrict__ src, const int* __restrict__ dst,
                            const float* __restrict__ dinv, const float* __restrict__ h,
                            float* __restrict__ out, int E) {
  const long long tid = (long long)blockIdx.x * blockDim.x + threadIdx.x;
  const long long e   = tid >> 3;
  const int lane      = (int)(tid & 7);
  if (e >= E) return;
  const int s = src[e], d = dst[e];
  const float nrm = dinv[s] * dinv[d];
  const float4 v = *(const float4*)(h + (size_t)s * GCN_CLS + lane * 4);
  float* o = out + (size_t)d * GCN_CLS + lane * 4;
  atomicAdd(o + 0, v.x * nrm);
  atomicAdd(o + 1, v.y * nrm);
  atomicAdd(o + 2, v.z * nrm);
  atomicAdd(o + 3, v.w * nrm);
}

// ---------------- self-loop + bias + ReLU + bf16 convert (layer 1) ----------------
__global__ void k_selfloop_bias_relu_bf16(const float* __restrict__ agg,
                                          const float* __restrict__ h,
                                          const float* __restrict__ dinv,
                                          const float* __restrict__ b,
                                          __bf16* __restrict__ outbf, size_t n) {
  size_t i = (size_t)blockIdx.x * blockDim.x + threadIdx.x;
  if (i >= n) return;
  const size_t node = i >> 7;  // /128
  const int    f    = (int)(i & 127);
  const float  di   = dinv[node];
  float t = agg[i] + h[i] * di * di + b[f];
  outbf[i] = (__bf16)fmaxf(t, 0.0f);
}

// ---------------- self-loop + bias (layer 2, final output) ----------------
__global__ void k_selfloop_bias(float* __restrict__ out, const float* __restrict__ h2,
                                const float* __restrict__ dinv,
                                const float* __restrict__ b, size_t n) {
  size_t i = (size_t)blockIdx.x * blockDim.x + threadIdx.x;
  if (i >= n) return;
  const size_t node = i >> 5;  // /32
  const int    f    = (int)(i & 31);
  const float  di   = dinv[node];
  out[i] = out[i] + h2[i] * di * di + b[f];
}

// ---------------------------------------------------------------------------
extern "C" void kernel_launch(void* const* d_in, const int* in_sizes, int n_in,
                              void* d_out, int out_size, void* d_ws, size_t ws_size,
                              hipStream_t stream) {
  (void)n_in; (void)ws_size;

  const float* x  = (const float*)d_in[0];
  const int*   ei = (const int*)d_in[1];
  const float* W1 = (const float*)d_in[2];
  const float* b1 = (const float*)d_in[3];
  const float* W2 = (const float*)d_in[4];
  const float* b2 = (const float*)d_in[5];
  float*       out = (float*)d_out;

  const int N = in_sizes[0] / GCN_IN;  // 100000
  const int E = in_sizes[1] / 2;       // 1600000
  const int* src = ei;                 // edge_index row 0
  const int* dst = ei + E;             // edge_index row 1

  // ---- carve workspace ----
  char*  ws  = (char*)d_ws;
  size_t off = 0;
  auto carve = [&](size_t bytes) -> void* {
    void* p = ws + off;
    off += (bytes + 255) & ~(size_t)255;
    return p;
  };
  __bf16*       xbf  = (__bf16*)carve((size_t)N * GCN_IN * 2);
  unsigned int* w1p  = (unsigned int*)carve((size_t)GCN_IN * GCN_HID * 2);   // packed
  unsigned int* w2p  = (unsigned int*)carve((size_t)GCN_HID * GCN_CLS * 2);  // packed
  float*        deg  = (float*)carve((size_t)N * 4);
  float*        dinv = (float*)carve((size_t)N * 4);
  float*        h1   = (float*)carve((size_t)N * GCN_HID * 4);
  float*        agg1 = (float*)carve((size_t)N * GCN_HID * 4);
  __bf16*       hbf  = (__bf16*)carve((size_t)N * GCN_HID * 2);
  float*        h2   = (float*)carve((size_t)N * GCN_CLS * 4);

  // ---- precision conversion + weight fragment packing ----
  k_f32_to_bf16<<<dim3(4096), dim3(256), 0, stream>>>(x, xbf, (size_t)N * GCN_IN);
  {
    int t1 = (GCN_IN / 32) * (GCN_HID / 16) * 256;  // packed dwords for W1
    k_pack_w<<<dim3((t1 + 255) / 256), dim3(256), 0, stream>>>(W1, w1p, GCN_IN, GCN_HID);
    int t2 = (GCN_HID / 32) * (GCN_CLS / 16) * 256;  // packed dwords for W2
    k_pack_w<<<dim3((t2 + 255) / 256), dim3(256), 0, stream>>>(W2, w2p, GCN_HID, GCN_CLS);
  }

  // ---- normalization: deg = 1 (self loop) + incoming-edge count; dinv ----
  k_fill_f32<<<dim3((N + 255) / 256), dim3(256), 0, stream>>>(deg, 1.0f, (size_t)N);
  k_degree<<<dim3((E + 255) / 256), dim3(256), 0, stream>>>(dst, deg, E);
  k_dinv<<<dim3((N + 255) / 256), dim3(256), 0, stream>>>(deg, dinv, N);

  // ---- layer 1: h1 = x @ W1 (WMMA bf16) ----
  const int mtiles = (N + 15) / 16;  // 6250 (exact)
  const int wavesPerBlock = 8;       // 256 threads
  dim3 gemmGrid((mtiles + wavesPerBlock - 1) / wavesPerBlock);
  dim3 gemmBlock(32 * wavesPerBlock);
  k_gemm_bf16<GCN_HID, GCN_IN><<<gemmGrid, gemmBlock, 0, stream>>>(
      xbf, (const v8u*)w1p, h1, N);

  // ---- layer 1 aggregation ----
  hipMemsetAsync(agg1, 0, (size_t)N * GCN_HID * 4, stream);
  {
    long long t = (long long)E * 32;
    k_scatter128<<<dim3((unsigned)((t + 255) / 256)), dim3(256), 0, stream>>>(
        src, dst, dinv, h1, agg1, E);
  }
  {
    size_t n = (size_t)N * GCN_HID;
    k_selfloop_bias_relu_bf16<<<dim3((unsigned)((n + 255) / 256)), dim3(256), 0, stream>>>(
        agg1, h1, dinv, b1, hbf, n);
  }

  // ---- layer 2: h2 = relu_h @ W2 (WMMA bf16) ----
  k_gemm_bf16<GCN_CLS, GCN_HID><<<gemmGrid, gemmBlock, 0, stream>>>(
      hbf, (const v8u*)w2p, h2, N);

  // ---- layer 2 aggregation into d_out ----
  hipMemsetAsync(out, 0, (size_t)out_size * 4, stream);
  {
    long long t = (long long)E * 8;
    k_scatter32<<<dim3((unsigned)((t + 255) / 256)), dim3(256), 0, stream>>>(
        src, dst, dinv, h2, out, E);
  }
  {
    size_t n = (size_t)N * GCN_CLS;
    k_selfloop_bias<<<dim3((unsigned)((n + 255) / 256)), dim3(256), 0, stream>>>(
        out, h2, dinv, b2, n);
  }
}
